// OneHeadAttention_79173427134567
// MI455X (gfx1250) — matrix operand order repaired and checked by reading, not compiled, not run on hardware
//
#include <hip/hip_runtime.h>
#include <hip/hip_bf16.h>

typedef __attribute__((ext_vector_type(16))) __bf16 v16bf;
typedef __attribute__((ext_vector_type(8)))  __bf16 v8bf;
typedef __attribute__((ext_vector_type(8)))  float  v8f;

#define B_ 4
#define S_ 4096
#define D_ 256
#define MS_ (B_ * S_)   // 16384 total rows

#define KCH   1024      // phase-3 K chunk (keys per LDS stage)
#define APAD  1032      // padded LDS row stride for attn chunk (2064 B, bank-spread)
#define CPAD  264       // padded LDS row stride for ctx tile   (528 B, bank-spread)

// ---- WMMA operand loaders (layouts per CDNA5 ISA 7.12.2, wave32) ----

// A-matrix 16x32 bf16: lane L -> row L%16; hf = L/16.
// element j<8  : K = hf*8 + j ;  element j>=8 : K = 16 + hf*8 + (j-8)
__device__ __forceinline__ v16bf a_from_bf16(const __bf16* row, int k0, int hf) {
    v8bf x0 = *(const v8bf*)(row + k0 + hf * 8);
    v8bf x1 = *(const v8bf*)(row + k0 + 16 + hf * 8);
    v16bf r;
#pragma unroll
    for (int i = 0; i < 8; ++i) { r[i] = x0[i]; r[i + 8] = x1[i]; }
    return r;
}

__device__ __forceinline__ v16bf a_from_f32(const float* row, int k0, int hf) {
    const float* p0 = row + k0 + hf * 8;
    const float* p1 = row + k0 + 16 + hf * 8;
    v16bf r;
#pragma unroll
    for (int i = 0; i < 8; ++i) { r[i] = (__bf16)p0[i]; r[i + 8] = (__bf16)p1[i]; }
    return r;
}

// B-matrix 32x16 bf16: lane L -> column L%16; hf = L/16.
// element e<8 : K = hf*8 + e ;  element e>=8 : K = 16 + hf*8 + (e-8)
// rowptr points into a [N x K] (transposed-B) bf16 array.
__device__ __forceinline__ v16bf b_row(const __bf16* rowptr, int k0, int hf) {
    v8bf x0 = *(const v8bf*)(rowptr + k0 + hf * 8);
    v8bf x1 = *(const v8bf*)(rowptr + k0 + 16 + hf * 8);
    v16bf r;
#pragma unroll
    for (int i = 0; i < 8; ++i) { r[i] = x0[i]; r[i + 8] = x1[i]; }
    return r;
}

__device__ __forceinline__ v8f wmma_bf16(v16bf a, v16bf b, v8f c) {
    return __builtin_amdgcn_wmma_f32_16x16x32_bf16(false, a, false, b, (short)0, c,
                                                   false, false);
}

// ---- Stage 0: W [K x N] f32  ->  WT [N x K] bf16 ----
__global__ void wtrans_kernel(const float* __restrict__ W, __bf16* __restrict__ WT) {
    int idx = blockIdx.x * blockDim.x + threadIdx.x;   // 0 .. 65535
    int n = idx / D_;
    int k = idx % D_;
    WT[idx] = (__bf16)W[k * D_ + n];
}

// ---- Stage 1: projections.  blockIdx.y selects q/k/v.  8 waves / block,
// each block computes one 16-row M tile across all 16 N tiles. ----
__global__ __launch_bounds__(256)
void proj_kernel(const float* __restrict__ q, const float* __restrict__ k,
                 const float* __restrict__ v,
                 const __bf16* __restrict__ WqT, const __bf16* __restrict__ WkT,
                 const __bf16* __restrict__ WvT,
                 const float* __restrict__ bq, const float* __restrict__ bk,
                 const float* __restrict__ bv,
                 __bf16* __restrict__ qp, __bf16* __restrict__ kp,
                 __bf16* __restrict__ vpT) {
    const int which = blockIdx.y;
    const float*  X    = (which == 0) ? q   : (which == 1) ? k   : v;
    const __bf16* WT   = (which == 0) ? WqT : (which == 1) ? WkT : WvT;
    const float*  bias = (which == 0) ? bq  : (which == 1) ? bk  : bv;

    const int m0   = blockIdx.x * 16;
    const int lane = threadIdx.x & 31;
    const int wave = threadIdx.x >> 5;
    const int hf   = lane >> 4;
    const int lc   = lane & 15;

    const float* xrow = X + (size_t)(m0 + lc) * D_;
    v16bf a[8];
#pragma unroll
    for (int kc = 0; kc < 8; ++kc) a[kc] = a_from_f32(xrow, kc * 32, hf);

#pragma unroll
    for (int t = 0; t < 2; ++t) {
        const int n0 = (wave * 2 + t) * 16;
        const __bf16* wrow = WT + (size_t)(n0 + lc) * D_;
        v8f c = {};
#pragma unroll
        for (int kc = 0; kc < 8; ++kc)
            c = wmma_bf16(a[kc], b_row(wrow, kc * 32, hf), c);

        const int n  = n0 + lc;
        const float bn = bias[n];
#pragma unroll
        for (int r = 0; r < 8; ++r) {
            const int m = m0 + hf * 8 + r;
            const float val = c[r] + bn;
            if (which == 0) {
                qp[(size_t)m * D_ + n] = (__bf16)val;
            } else if (which == 1) {
                kp[(size_t)m * D_ + n] = (__bf16)val;
            } else {
                const int bb = m / S_;
                const int s  = m % S_;
                vpT[((size_t)bb * D_ + n) * (size_t)S_ + s] = (__bf16)val;
            }
        }
    }
}

// ---- Stage 2: fused logits -> softmax -> ctx -> output projection.
// One block per 16-query tile (grid = B*S/16 = 1024), 256 threads = 8 waves. ----
__global__ __launch_bounds__(256)
void attn_kernel(const __bf16* __restrict__ qp, const __bf16* __restrict__ kp,
                 const __bf16* __restrict__ vpT, const float* __restrict__ mask,
                 const __bf16* __restrict__ WoT, const float* __restrict__ bo,
                 float* __restrict__ z, float* __restrict__ attn) {
    __shared__ __bf16 attn_lds[16 * APAD];  // 33 KB: one 16x1024 prob chunk, bf16
    __shared__ __bf16 ctx_lds[16 * CPAD];   // 8.25 KB
    __shared__ float  red[16 * 16];
    __shared__ float  rowstat[16];

    const int m0   = blockIdx.x * 16;       // global query row over B*S
    const int bb   = m0 / S_;
    const int q0   = m0 % S_;               // query row within batch
    const int lane = threadIdx.x & 31;
    const int wave = threadIdx.x >> 5;
    const int hf   = lane >> 4;
    const int lc   = lane & 15;

    // A-regs: this block's 16x256 qp tile (whole K held in registers).
    const __bf16* qrow = qp + (size_t)(m0 + lc) * D_;
    v16bf aq[8];
#pragma unroll
    for (int kc = 0; kc < 8; ++kc) aq[kc] = a_from_bf16(qrow, kc * 32, hf);

    const float scale = 0.0625f;            // 1/sqrt(256)
    const size_t attn_base = (size_t)m0 * S_;

    // ---- Phase 1: logits for the 16 x 4096 strip; wave w does key tiles w, w+8, ...
    for (int kt = wave; kt < S_ / 16; kt += 8) {
        const int n0 = kt * 16;
        const __bf16* krow = kp + ((size_t)(bb * S_ + n0 + lc)) * D_;
        if (kt + 8 < S_ / 16)                       // prefetch next tile's K rows
            __builtin_prefetch((const void*)(krow + (size_t)128 * D_), 0, 3);
        v8f c = {};
#pragma unroll
        for (int kc = 0; kc < 8; ++kc)
            c = wmma_bf16(aq[kc], b_row(krow, kc * 32, hf), c);

        const int n = n0 + lc;
#pragma unroll
        for (int r = 0; r < 8; ++r) {
            const int ml = hf * 8 + r;
            const float mval = mask[(size_t)(q0 + ml) * S_ + n];
            attn[attn_base + (size_t)ml * S_ + n] = c[r] * scale + mval * -1e9f;
        }
    }
    __threadfence();
    __syncthreads();

    // ---- Phase 2: row max + exp + row sum (strip lives in L2).
    const int row = threadIdx.x >> 4;   // 0..15
    const int sub = threadIdx.x & 15;   // 0..15
    {
        float* arow = attn + attn_base + (size_t)row * S_;

        float mx = -3.4e38f;
        for (int j = sub; j < S_; j += 16) mx = fmaxf(mx, arow[j]);
        red[row * 16 + sub] = mx;
        __syncthreads();
        if (sub == 0) {
            float m2 = red[row * 16];
            for (int i = 1; i < 16; ++i) m2 = fmaxf(m2, red[row * 16 + i]);
            rowstat[row] = m2;
        }
        __syncthreads();
        const float rmax = rowstat[row];

        float sum = 0.f;
        for (int j = sub; j < S_; j += 16) {
            const float e = __expf(arow[j] - rmax);
            arow[j] = e;                 // unnormalized; each thread re-reads only
            sum += e;                    // its own elements in phase 3 staging
        }
        red[row * 16 + sub] = sum;
        __syncthreads();
        if (sub == 0) {
            float s2 = 0.f;
            for (int i = 0; i < 16; ++i) s2 += red[row * 16 + i];
            rowstat[row] = 1.0f / s2;
        }
        __syncthreads();
    }

    // ---- Phase 3: ctx = attn @ vp, K = 4096 chunked by 1024 through LDS.
    // Staging pass normalizes (write final attn f32) and converts to bf16 once;
    // all 8 waves then feed WMMA A-operands from LDS.
    const float inv = rowstat[row];
    const int   n0v = wave * 32;
    const __bf16* vrow0 = vpT + ((size_t)bb * D_ + n0v + lc) * (size_t)S_;
    const __bf16* vrow1 = vrow0 + (size_t)16 * S_;
    v8f c0 = {}, c1 = {};

    for (int ch = 0; ch < S_ / KCH; ++ch) {
        // stage chunk [16 x KCH]: normalize + convert into LDS
        float*  arow = attn + attn_base + (size_t)row * S_ + ch * KCH;
        __bf16* lrow = attn_lds + row * APAD;
#pragma unroll 4
        for (int i = 0; i < KCH / 16; ++i) {
            const int j = sub + 16 * i;
            const float p = arow[j] * inv;   // own earlier write (exp value)
            arow[j] = p;                      // final attn output
            lrow[j] = (__bf16)p;
        }
        __syncthreads();

        const __bf16* alds = attn_lds + lc * APAD;
        const int kbase = ch * KCH;
        for (int kc = 0; kc < KCH / 32; ++kc) {
            v16bf a = a_from_bf16(alds, kc * 32, hf);
            c0 = wmma_bf16(a, b_row(vrow0, kbase + kc * 32, hf), c0);
            c1 = wmma_bf16(a, b_row(vrow1, kbase + kc * 32, hf), c1);
        }
        __syncthreads();
    }

#pragma unroll
    for (int r = 0; r < 8; ++r) {
        const int ml = hf * 8 + r;
        ctx_lds[ml * CPAD + n0v + lc]      = (__bf16)c0[r];
        ctx_lds[ml * CPAD + n0v + 16 + lc] = (__bf16)c1[r];
    }
    __syncthreads();

    // ---- Phase 4: z = ctx @ WoT + bo.
    {
        const __bf16* crow = ctx_lds + lc * CPAD;
        v16bf ac[8];
#pragma unroll
        for (int kc = 0; kc < 8; ++kc) ac[kc] = a_from_bf16(crow, kc * 32, hf);

#pragma unroll
        for (int t = 0; t < 2; ++t) {
            const int n0 = (wave * 2 + t) * 16;
            const __bf16* worow = WoT + (size_t)(n0 + lc) * D_;
            v8f c = {};
#pragma unroll
            for (int kc = 0; kc < 8; ++kc)
                c = wmma_bf16(ac[kc], b_row(worow, kc * 32, hf), c);

            const int n  = n0 + lc;
            const float bn = bo[n];
#pragma unroll
            for (int r = 0; r < 8; ++r)
                z[(size_t)(m0 + hf * 8 + r) * D_ + n] = c[r] + bn;
        }
    }
}

extern "C" void kernel_launch(void* const* d_in, const int* in_sizes, int n_in,
                              void* d_out, int out_size, void* d_ws, size_t ws_size,
                              hipStream_t stream) {
    (void)in_sizes; (void)n_in; (void)out_size; (void)ws_size;

    const float* q    = (const float*)d_in[0];
    const float* k    = (const float*)d_in[1];
    const float* v    = (const float*)d_in[2];
    const float* mask = (const float*)d_in[3];
    const float* Wq   = (const float*)d_in[4];
    const float* bq   = (const float*)d_in[5];
    const float* Wk   = (const float*)d_in[6];
    const float* bk   = (const float*)d_in[7];
    const float* Wv   = (const float*)d_in[8];
    const float* bv   = (const float*)d_in[9];
    const float* Wo   = (const float*)d_in[10];
    const float* bo   = (const float*)d_in[11];

    float* z    = (float*)d_out;                       // [B,S,D]
    float* attn = z + (size_t)B_ * S_ * D_;            // [B,S,S]

    // Workspace (bf16): qp, kp row-major [B*S, D]; vpT [B, D, S]; 4x WT [D, D].
    char* ws = (char*)d_ws;
    __bf16* qp  = (__bf16*)ws;  ws += (size_t)MS_ * D_ * sizeof(__bf16);
    __bf16* kp  = (__bf16*)ws;  ws += (size_t)MS_ * D_ * sizeof(__bf16);
    __bf16* vpT = (__bf16*)ws;  ws += (size_t)MS_ * D_ * sizeof(__bf16);
    __bf16* WqT = (__bf16*)ws;  ws += (size_t)D_ * D_ * sizeof(__bf16);
    __bf16* WkT = (__bf16*)ws;  ws += (size_t)D_ * D_ * sizeof(__bf16);
    __bf16* WvT = (__bf16*)ws;  ws += (size_t)D_ * D_ * sizeof(__bf16);
    __bf16* WoT = (__bf16*)ws;  ws += (size_t)D_ * D_ * sizeof(__bf16);

    wtrans_kernel<<<(D_ * D_) / 256, 256, 0, stream>>>(Wq, WqT);
    wtrans_kernel<<<(D_ * D_) / 256, 256, 0, stream>>>(Wk, WkT);
    wtrans_kernel<<<(D_ * D_) / 256, 256, 0, stream>>>(Wv, WvT);
    wtrans_kernel<<<(D_ * D_) / 256, 256, 0, stream>>>(Wo, WoT);

    proj_kernel<<<dim3(MS_ / 16, 3), 256, 0, stream>>>(q, k, v, WqT, WkT, WvT,
                                                       bq, bk, bv, qp, kp, vpT);

    attn_kernel<<<MS_ / 16, 256, 0, stream>>>(qp, kp, vpT, mask, WoT, bo, z, attn);
}
